// MotionTransformerAttentionLayer_25821343384048
// MI455X (gfx1250) — matrix operand order repaired and checked by reading, not compiled, not run on hardware
//
#include <hip/hip_runtime.h>

typedef _Float16 f16;
typedef _Float16 v16h __attribute__((ext_vector_type(16)));
typedef _Float16 v8h  __attribute__((ext_vector_type(8)));
typedef float    v8f  __attribute__((ext_vector_type(8)));
typedef unsigned int u32x4 __attribute__((ext_vector_type(4)));
typedef int          i32x4 __attribute__((ext_vector_type(4)));
typedef int          i32x8 __attribute__((ext_vector_type(8)));

// ---- problem dims (fixed by reference setup_inputs) ----
#define BS    4
#define NA    256
#define NM    6
#define NQ    (NA*NM)      // 1536
#define EDIM  256
#define NH    8
#define DH    32           // EDIM/NH
#define NS    12
#define NP    4
#define NV    40000        // 200*200
#define GRID_HW 200
#define NOFF  768          // nh*ns*nl*np*2
#define NAW   384          // nh*ns*nl*np
#define KOUT  3072         // ns*EDIM
#define NROW  (BS*NQ)      // 6144

// LDS tile row stride in halves: 32 data + 8 pad (= TDM pad 4 DWORDs / 16 DWORD interval)
#define LDSS  40

// Build TDM D# groups for a 2D f16 tile (tileK x tileN) from a (rows x K) f16
// tensor, row stride = strideK elements, destination LDS with 8-half row pad.
__device__ __forceinline__ void tdm_load_tile_f16(unsigned ldsOff, const f16* gptr,
                                                  int tileK, int tileN,
                                                  int tensorK, int tensorN, int strideK)
{
    unsigned long long ga = (unsigned long long)(size_t)gptr;
    u32x4 g0;
    g0[0] = 1u;                                            // count=1, user mode, no gather
    g0[1] = ldsOff;                                        // lds_addr (bytes)
    g0[2] = (unsigned)(ga & 0xffffffffu);                  // global_addr[31:0]
    g0[3] = (unsigned)((ga >> 32) & 0x01ffffffu)           // global_addr[56:32]
          | (2u << 30);                                    // type=2 ("image")
    i32x8 g1;
    g1[0] = (1 << 16)                                      // data_size: 2 bytes
          | (1 << 20)                                      // pad_enable
          | (3 << 22)                                      // pad_interval: 16 DWORDs (=32 halves)
          | (3 << 25);                                     // pad_amount:   4 DWORDs (= 8 halves)
    g1[1] = (tensorK & 0xffff) << 16;                      // tensor_dim0[15:0]
    g1[2] = ((tensorK >> 16) & 0xffff)                     // tensor_dim0[31:16]
          | ((tensorN & 0xffff) << 16);                    // tensor_dim1[15:0]
    g1[3] = ((tensorN >> 16) & 0xffff)                     // tensor_dim1[31:16]
          | ((tileK & 0xffff) << 16);                      // tile_dim0
    g1[4] = (tileN & 0xffff);                              // tile_dim1 (tile_dim2 = 0)
    g1[5] = strideK;                                       // tensor_dim0_stride[31:0]
    g1[6] = 0;                                             // stride0 hi | stride1 lo
    g1[7] = 0;                                             // stride1 hi
    i32x4 gz4 = {0, 0, 0, 0};
    i32x8 gz8 = {0, 0, 0, 0, 0, 0, 0, 0};
    // clang-23 / therock-10.0 6-arg form: (g0, g1, g2, g3, g4, cpol)
    __builtin_amdgcn_tensor_load_to_lds(g0, g1, gz4, gz4, gz8, 0);
}

// =====================================================================
// WMMA GEMM:  C[z] = A[z] (MxK) @ Bt^T (Bt is N x K, f16) + bias
// Block tile 64(M) x 128(N), 8 waves, wave tile 32x32 (4x v_wmma 16x16x32)
// All launch dims are exact multiples of the tiles -> no bounds guards.
// B tile (and A tile when already f16) staged by the Tensor Data Mover.
// =====================================================================
template<bool A_F32, bool OUT_F16>
__global__ __launch_bounds__(256)
void gemm_wmma(const void* __restrict__ Aptr, long aBatch, int lda,
               const f16* __restrict__ Bt, const float* __restrict__ bias,
               void* __restrict__ Cptr, long cBatch,
               int M, int N, int K)
{
    __shared__ f16 sA[64 * LDSS];    // 64 rows x 32 k (+pad)
    __shared__ f16 sB[128 * LDSS];   // 128 cols x 32 k (+pad)

    const int tid  = threadIdx.x;
    const int lane = tid & 31;
    const int wid  = tid >> 5;
    const int wm   = (wid & 1) * 32;   // wave M offset in block tile
    const int wn   = (wid >> 1) * 32;  // wave N offset in block tile
    const int bm   = blockIdx.y * 64;
    const int bn   = blockIdx.x * 128;
    const int z    = blockIdx.z;

    const float* Af = (const float*)Aptr + (A_F32 ? (long)z * aBatch : 0);
    const f16*   Ah = (const f16*)Aptr   + (A_F32 ? 0 : (long)z * aBatch);

    v8f c00 = {}, c01 = {}, c10 = {}, c11 = {};

    // A staging decomposition (f32 path only)
    const int ar = tid >> 2;          // 0..63
    const int ak = (tid & 3) * 8;     // 0,8,16,24

    const int lrow  = lane & 15;
    const int hi    = lane >> 4;      // 0 | 1
    const int akoff = hi * 8;         // A frag K-half select
    const int bkoff = hi * 8;         // B frag K-half select

    const unsigned sAoff = (unsigned)(size_t)(&sA[0]);   // LDS byte offset (low 32 bits of flat)
    const unsigned sBoff = (unsigned)(size_t)(&sB[0]);

    for (int kb = 0; kb < K; kb += 32) {
        // ---- stage tiles ----
        if (A_F32) {
            // per-thread convert f32 -> f16 into LDS
            const float* src = Af + (long)(bm + ar) * lda + kb + ak;
            f16* dst = &sA[ar * LDSS + ak];
            #pragma unroll
            for (int i = 0; i < 8; ++i) dst[i] = (f16)src[i];
            __builtin_prefetch(src + 32, 0, 1);
        }
        if (wid == 0) {
            // Tensor Data Mover: B tile (32 k x 128 n), padded rows -> stride 40 halves
            tdm_load_tile_f16(sBoff, Bt + (long)bn * K + kb, 32, 128, K, N, K);
            if (!A_F32) {
                // A tile (32 k x 64 rows) straight from f16 A
                tdm_load_tile_f16(sAoff, Ah + (long)bm * lda + kb, 32, 64, K, M, lda);
            }
            __builtin_amdgcn_s_wait_tensorcnt(0);
        }
        __syncthreads();

        // ---- load fragments per ISA VGPR layout, 4x wmma ----
        v16h a0, a1, b0, b1;
        #pragma unroll
        for (int i = 0; i < 8; ++i) {
            a0[i]     = sA[(wm + lrow) * LDSS + akoff + i];
            a0[8 + i] = sA[(wm + lrow) * LDSS + 16 + akoff + i];
            a1[i]     = sA[(wm + 16 + lrow) * LDSS + akoff + i];
            a1[8 + i] = sA[(wm + 16 + lrow) * LDSS + 16 + akoff + i];
        }
        #pragma unroll
        for (int i = 0; i < 16; ++i) {
            b0[i] = sB[(wn + lrow) * LDSS + bkoff + i + (i >= 8 ? 8 : 0)];
            b1[i] = sB[(wn + 16 + lrow) * LDSS + bkoff + i + (i >= 8 ? 8 : 0)];
        }
        c00 = __builtin_amdgcn_wmma_f32_16x16x32_f16(false, a0, false, b0, (short)0, c00, false, false);
        c01 = __builtin_amdgcn_wmma_f32_16x16x32_f16(false, a0, false, b1, (short)0, c01, false, false);
        c10 = __builtin_amdgcn_wmma_f32_16x16x32_f16(false, a1, false, b0, (short)0, c10, false, false);
        c11 = __builtin_amdgcn_wmma_f32_16x16x32_f16(false, a1, false, b1, (short)0, c11, false, false);
        __syncthreads();
    }

    // ---- epilogue: bias + store (C layout: VGPR r -> row r + 8*hi, col = lane&15) ----
    const int rowBase = bm + wm + hi * 8;
    const int col0 = bn + wn + lrow;
    const int col1 = col0 + 16;
    const float bv0 = bias[col0];
    const float bv1 = bias[col1];

    f16*   Ch = (f16*)Cptr   + (OUT_F16 ? (long)z * cBatch : 0);
    float* Cf = (float*)Cptr + (OUT_F16 ? 0 : (long)z * cBatch);

    #pragma unroll
    for (int r = 0; r < 8; ++r) {
        const int m0 = rowBase + r;        // frag fm=0
        const int m1 = rowBase + 16 + r;   // frag fm=1
        if (OUT_F16) {
            Ch[(long)m0 * N + col0] = (f16)(c00[r] + bv0);
            Ch[(long)m0 * N + col1] = (f16)(c01[r] + bv1);
            Ch[(long)m1 * N + col0] = (f16)(c10[r] + bv0);
            Ch[(long)m1 * N + col1] = (f16)(c11[r] + bv1);
        } else {
            Cf[(long)m0 * N + col0] = c00[r] + bv0;
            Cf[(long)m0 * N + col1] = c01[r] + bv1;
            Cf[(long)m1 * N + col0] = c10[r] + bv0;
            Cf[(long)m1 * N + col1] = c11[r] + bv1;
        }
    }
}

// =====================================================================
// q = query + query_pos (f32 elementwise)
// =====================================================================
__global__ __launch_bounds__(256)
void qprep(const float* __restrict__ a, const float* __restrict__ b,
           float* __restrict__ c, int n)
{
    int i = blockIdx.x * 256 + threadIdx.x;
    if (i < n) c[i] = a[i] + b[i];
}

// =====================================================================
// Wt[n*K + k] = (f16) W[k*N + n]   (weights K x N  ->  f16 N x K)
// =====================================================================
__global__ __launch_bounds__(256)
void wtrans(const float* __restrict__ W, f16* __restrict__ Wt, int K, int N)
{
    long i = (long)blockIdx.x * 256 + threadIdx.x;
    if (i < (long)K * N) {
        int n = (int)(i / K), k = (int)(i % K);
        Wt[i] = (f16)W[(long)k * N + n];
    }
}

// =====================================================================
// MSDA sampler: one wave per (b,q,s,h); lane = d (0..31).
// Fused softmax over 4 points + bilinear gather from f16 vproj
// (vproj = 82 MB f16 -> resident in the 192MB L2; gathers run at L2 BW).
// =====================================================================
__global__ __launch_bounds__(256)
void msda_sample(const float* __restrict__ offb, const float* __restrict__ awb,
                 const float* __restrict__ ref_trajs, const float* __restrict__ det_centers,
                 const f16* __restrict__ vproj, f16* __restrict__ sampled)
{
    const int w    = blockIdx.x * 8 + (threadIdx.x >> 5);
    const int lane = threadIdx.x & 31;
    const int h = w & 7;
    const int s = (w >> 3) % NS;
    const int q = (w / (8 * NS)) % NQ;
    const int b = w / (8 * NS * NQ);

    // reference point (agent frame + center -> normalized BEV)
    const int a = q / NM, m = q % NM;
    const long rtBase = ((((long)(b * NA + a) * NM + m) * NS + (NS - 1))) * 2;
    const float rtx = ref_trajs[rtBase]     + det_centers[(b * NA + a) * 2 + 0];
    const float rty = ref_trajs[rtBase + 1] + det_centers[(b * NA + a) * 2 + 1];
    const float rx = (rtx + 51.2f) * (1.f / 102.4f);
    const float ry = (rty + 51.2f) * (1.f / 102.4f);

    // softmax over the 4 points
    const long awBase = (long)(b * NQ + q) * NAW + h * (NS * NP) + s * NP;
    const float w0 = awb[awBase], w1 = awb[awBase + 1], w2 = awb[awBase + 2], w3 = awb[awBase + 3];
    const float mx = fmaxf(fmaxf(w0, w1), fmaxf(w2, w3));
    const float e0 = expf(w0 - mx), e1 = expf(w1 - mx), e2 = expf(w2 - mx), e3 = expf(w3 - mx);
    const float inv = 1.f / (e0 + e1 + e2 + e3);
    const float wp[4] = {e0 * inv, e1 * inv, e2 * inv, e3 * inv};

    const long offBase = (long)(b * NQ + q) * NOFF + h * (NS * NP * 2) + s * (NP * 2);
    const f16* vb = vproj + (((long)b * NV) * NH + h) * DH + lane;  // + idx*256

    float acc = 0.f;
    #pragma unroll
    for (int p = 0; p < NP; ++p) {
        const float px = rx * (float)GRID_HW + offb[offBase + p * 2]     - 0.5f;
        const float py = ry * (float)GRID_HW + offb[offBase + p * 2 + 1] - 0.5f;
        const float x0 = floorf(px), y0 = floorf(py);
        const float fx = px - x0, fy = py - y0;
        const int ix0 = (int)x0, iy0 = (int)y0;
        const float wgt = wp[p];
        #pragma unroll
        for (int dy = 0; dy < 2; ++dy) {
            #pragma unroll
            for (int dx = 0; dx < 2; ++dx) {
                const int xi = ix0 + dx, yi = iy0 + dy;
                const float wc = (dx ? fx : 1.f - fx) * (dy ? fy : 1.f - fy);
                if (xi >= 0 && xi < GRID_HW && yi >= 0 && yi < GRID_HW) {
                    const long idx = (long)yi * GRID_HW + xi;
                    acc += wgt * wc * (float)vb[idx * (NH * DH)];
                }
            }
        }
    }
    // layout for final GEMM A: row (b*NQ+q), col s*EDIM + h*DH + d
    sampled[(long)(b * NQ + q) * KOUT + s * EDIM + h * DH + lane] = (f16)acc;
}

// =====================================================================
// LayerNorm + ReLU + residual: one wave per row of 256.
// =====================================================================
__global__ __launch_bounds__(256)
void ln_relu_res(const float* __restrict__ out2, const float* __restrict__ query,
                 const float* __restrict__ g, const float* __restrict__ beta,
                 float* __restrict__ out)
{
    const int row  = blockIdx.x * 8 + (threadIdx.x >> 5);
    const int lane = threadIdx.x & 31;
    const float* x = out2 + (long)row * EDIM;

    float v[8], sum = 0.f, sq = 0.f;
    #pragma unroll
    for (int i = 0; i < 8; ++i) {
        v[i] = x[i * 32 + lane];
        sum += v[i];
        sq  += v[i] * v[i];
    }
    #pragma unroll
    for (int msk = 16; msk > 0; msk >>= 1) {
        sum += __shfl_xor(sum, msk, 32);
        sq  += __shfl_xor(sq,  msk, 32);
    }
    const float mu  = sum * (1.f / EDIM);
    const float var = sq * (1.f / EDIM) - mu * mu;
    const float rstd = rsqrtf(var + 1e-5f);
    #pragma unroll
    for (int i = 0; i < 8; ++i) {
        const int e = i * 32 + lane;
        float y = (v[i] - mu) * rstd * g[e] + beta[e];
        y = fmaxf(y, 0.f);
        out[(long)row * EDIM + e] = y + query[(long)row * EDIM + e];
    }
}

// =====================================================================
extern "C" void kernel_launch(void* const* d_in, const int* in_sizes, int n_in,
                              void* d_out, int out_size, void* d_ws, size_t ws_size,
                              hipStream_t stream)
{
    const float* query      = (const float*)d_in[0];
    const float* query_pos  = (const float*)d_in[1];
    const float* value      = (const float*)d_in[2];
    const float* ref_trajs  = (const float*)d_in[3];
    const float* det_cent   = (const float*)d_in[4];
    const float* W_value    = (const float*)d_in[5];
    const float* b_value    = (const float*)d_in[6];
    const float* W_off      = (const float*)d_in[7];
    const float* b_off      = (const float*)d_in[8];
    const float* W_attn     = (const float*)d_in[9];
    const float* b_attn     = (const float*)d_in[10];
    const float* W_out      = (const float*)d_in[11];
    const float* b_out      = (const float*)d_in[12];
    const float* ln_g       = (const float*)d_in[13];
    const float* ln_b       = (const float*)d_in[14];

    // ---- workspace carve-out ----
    char* wsp = (char*)d_ws;
    size_t o = 0;
    auto alloc = [&](size_t bytes) -> void* {
        void* p = wsp + o;
        o = (o + bytes + 255) & ~(size_t)255;
        return p;
    };
    float* qf32    = (float*)alloc((size_t)NROW * EDIM * 4);          // 6.3 MB
    f16*   Wv_t    = (f16*)  alloc((size_t)EDIM * EDIM * 2);
    f16*   Woff_t  = (f16*)  alloc((size_t)EDIM * NOFF * 2);
    f16*   Wattn_t = (f16*)  alloc((size_t)EDIM * NAW * 2);
    f16*   Wout_t  = (f16*)  alloc((size_t)KOUT * EDIM * 2);
    f16*   vproj   = (f16*)  alloc((size_t)BS * NV * EDIM * 2);       // 82 MB (fits L2)
    float* offbuf  = (float*)alloc((size_t)NROW * NOFF * 4);          // 18.9 MB
    float* awbuf   = (float*)alloc((size_t)NROW * NAW * 4);           // 9.4 MB
    f16*   sampled = (f16*)  alloc((size_t)NROW * KOUT * 2);          // 37.7 MB
    float* out2    = (float*)alloc((size_t)NROW * EDIM * 4);          // 6.3 MB
    (void)ws_size; (void)in_sizes; (void)n_in; (void)out_size;

    // 1) q = query + query_pos
    qprep<<<(NROW * EDIM + 255) / 256, 256, 0, stream>>>(query, query_pos, qf32, NROW * EDIM);

    // 2) weight transposes -> f16 N x K
    wtrans<<<(EDIM * EDIM + 255) / 256, 256, 0, stream>>>(W_value, Wv_t, EDIM, EDIM);
    wtrans<<<(EDIM * NOFF + 255) / 256, 256, 0, stream>>>(W_off,  Woff_t, EDIM, NOFF);
    wtrans<<<(EDIM * NAW  + 255) / 256, 256, 0, stream>>>(W_attn, Wattn_t, EDIM, NAW);
    wtrans<<<(KOUT * EDIM + 255) / 256, 256, 0, stream>>>(W_out,  Wout_t, KOUT, EDIM);

    // 3) value projection: per batch b, A row m at value[m*(BS*E) + b*E], out f16 (b,nv,h,d)
    {
        dim3 grid(EDIM / 128, NV / 64, BS);
        gemm_wmma<true, true><<<grid, 256, 0, stream>>>(
            (const void*)value, (long)EDIM, BS * EDIM, Wv_t, b_value,
            (void*)vproj, (long)NV * EDIM, NV, EDIM, EDIM);
    }
    // 4) offsets: (6144 x 256) @ (256 x 768)
    {
        dim3 grid(NOFF / 128, NROW / 64, 1);
        gemm_wmma<true, false><<<grid, 256, 0, stream>>>(
            (const void*)qf32, 0L, EDIM, Woff_t, b_off,
            (void*)offbuf, 0L, NROW, NOFF, EDIM);
    }
    // 5) attention weights: (6144 x 256) @ (256 x 384)
    {
        dim3 grid(NAW / 128, NROW / 64, 1);
        gemm_wmma<true, false><<<grid, 256, 0, stream>>>(
            (const void*)qf32, 0L, EDIM, Wattn_t, b_attn,
            (void*)awbuf, 0L, NROW, NAW, EDIM);
    }
    // 6) MSDA bilinear sampling (fused softmax); 1 wave per (b,q,s,h)
    {
        const int nwaves = BS * NQ * NS * NH;   // 589824
        msda_sample<<<nwaves / 8, 256, 0, stream>>>(offbuf, awbuf, ref_trajs, det_cent, vproj, sampled);
    }
    // 7) output projection: (6144 x 3072) @ (3072 x 256), A already f16 -> TDM-staged
    {
        dim3 grid(EDIM / 128, NROW / 64, 1);
        gemm_wmma<false, false><<<grid, 256, 0, stream>>>(
            (const void*)sampled, 0L, KOUT, Wout_t, b_out,
            (void*)out2, 0L, NROW, EDIM, KOUT);
    }
    // 8) LayerNorm + ReLU + residual -> d_out
    ln_relu_res<<<NROW / 8, 256, 0, stream>>>(out2, query, ln_g, ln_b, (float*)d_out);
}
